// AdaPT_Linear_Brevitas_87067577025510
// MI455X (gfx1250) — 8-bit matrix operand order repaired and checked by reading, compiled, not run on hardware
//
#include <hip/hip_runtime.h>


typedef int v8i __attribute__((ext_vector_type(8)));

#define K_DIM  1024
#define M_OUT  1024
#define QMAXF  127.0f

__global__ void init_slots(unsigned* __restrict__ slots) {
    if (threadIdx.x < 2) slots[threadIdx.x] = 0u;
}

__global__ __launch_bounds__(256)
void absmax_kernel(const float* __restrict__ in, long long n4,
                   unsigned* __restrict__ slot) {
    const float4* in4 = (const float4*)in;
    float m = 0.0f;
    long long i = (long long)blockIdx.x * blockDim.x + threadIdx.x;
    const long long stride = (long long)gridDim.x * blockDim.x;
    for (; i < n4; i += stride) {
        float4 v = in4[i];
        m = fmaxf(m, fmaxf(fmaxf(fabsf(v.x), fabsf(v.y)),
                           fmaxf(fabsf(v.z), fabsf(v.w))));
    }
    __shared__ float red[256];
    red[threadIdx.x] = m;
    __syncthreads();
    for (int s = 128; s > 0; s >>= 1) {
        if (threadIdx.x < (unsigned)s)
            red[threadIdx.x] = fmaxf(red[threadIdx.x], red[threadIdx.x + s]);
        __syncthreads();
    }
    if (threadIdx.x == 0) atomicMax(slot, __float_as_uint(red[0]));
}

__global__ __launch_bounds__(256)
void quantize_kernel(const float* __restrict__ in, int* __restrict__ out,
                     long long n4, const unsigned* __restrict__ slot) {
    const float inv = QMAXF / __uint_as_float(*slot);
    const float4* in4 = (const float4*)in;
    long long i = (long long)blockIdx.x * blockDim.x + threadIdx.x;
    const long long stride = (long long)gridDim.x * blockDim.x;
    for (; i < n4; i += stride) {
        float4 v = in4[i];
        int a = __float2int_rn(fminf(fmaxf(v.x * inv, -QMAXF), QMAXF));
        int b = __float2int_rn(fminf(fmaxf(v.y * inv, -QMAXF), QMAXF));
        int c = __float2int_rn(fminf(fmaxf(v.z * inv, -QMAXF), QMAXF));
        int d = __float2int_rn(fminf(fmaxf(v.w * inv, -QMAXF), QMAXF));
        out[i] = (a & 255) | ((b & 255) << 8) | ((c & 255) << 16) | ((d & 255) << 24);
    }
}

__global__ __launch_bounds__(256)
void gemm_i8_wmma(const int* __restrict__ qx, const int* __restrict__ qw,
                  const float* __restrict__ bias, float* __restrict__ out,
                  const unsigned* __restrict__ slots) {
    const int lane = threadIdx.x & 31;
    const int wave = threadIdx.x >> 5;
    const int wm = wave & 3;
    const int wn = wave >> 2;
    const int m0 = blockIdx.x * 128 + wm * 32;
    const int n0 = blockIdx.y * 64  + wn * 32;

    const int half = lane >> 4;
    const int l16  = lane & 15;

    const int Kd = K_DIM / 4;

    const int* __restrict__ aptr0 = qx + (long long)(m0 + l16)       * Kd;
    const int* __restrict__ aptr1 = qx + (long long)(m0 + 16 + l16)  * Kd;
    const int* __restrict__ bptr0 = qw + (long long)(n0 + l16)       * Kd;
    const int* __restrict__ bptr1 = qw + (long long)(n0 + 16 + l16)  * Kd;

    const int ab = half * 2;
    const int bb = half * 2;

    v8i c00 = {}, c01 = {}, c10 = {}, c11 = {};

    for (int k = 0; k < K_DIM; k += 64) {
        const int kd = k >> 2;

        __builtin_prefetch(aptr0 + kd + 16, 0, 1);
        __builtin_prefetch(aptr1 + kd + 16, 0, 1);

        v8i a0, a1, b0, b1;
        a0[0] = aptr0[kd + ab + 0];  a0[1] = aptr0[kd + ab + 1];
        a0[2] = aptr0[kd + ab + 4];  a0[3] = aptr0[kd + ab + 5];
        a0[4] = aptr0[kd + ab + 8];  a0[5] = aptr0[kd + ab + 9];
        a0[6] = aptr0[kd + ab + 12]; a0[7] = aptr0[kd + ab + 13];

        a1[0] = aptr1[kd + ab + 0];  a1[1] = aptr1[kd + ab + 1];
        a1[2] = aptr1[kd + ab + 4];  a1[3] = aptr1[kd + ab + 5];
        a1[4] = aptr1[kd + ab + 8];  a1[5] = aptr1[kd + ab + 9];
        a1[6] = aptr1[kd + ab + 12]; a1[7] = aptr1[kd + ab + 13];

        b0[0] = bptr0[kd + bb + 0];  b0[1] = bptr0[kd + bb + 1];
        b0[2] = bptr0[kd + bb + 4];  b0[3] = bptr0[kd + bb + 5];
        b0[4] = bptr0[kd + bb + 8];  b0[5] = bptr0[kd + bb + 9];
        b0[6] = bptr0[kd + bb + 12]; b0[7] = bptr0[kd + bb + 13];

        b1[0] = bptr1[kd + bb + 0];  b1[1] = bptr1[kd + bb + 1];
        b1[2] = bptr1[kd + bb + 4];  b1[3] = bptr1[kd + bb + 5];
        b1[4] = bptr1[kd + bb + 8];  b1[5] = bptr1[kd + bb + 9];
        b1[6] = bptr1[kd + bb + 12]; b1[7] = bptr1[kd + bb + 13];

        c00 = __builtin_amdgcn_wmma_i32_16x16x64_iu8(true, a0, true, b0, c00, false, false);
        c01 = __builtin_amdgcn_wmma_i32_16x16x64_iu8(true, a0, true, b1, c01, false, false);
        c10 = __builtin_amdgcn_wmma_i32_16x16x64_iu8(true, a1, true, b0, c10, false, false);
        c11 = __builtin_amdgcn_wmma_i32_16x16x64_iu8(true, a1, true, b1, c11, false, false);
    }

    const float sx = __uint_as_float(slots[0]) / QMAXF;
    const float sw = __uint_as_float(slots[1]) / QMAXF;
    const float s  = sx * sw;

    const int nA = n0 + l16;
    const int nB = n0 + 16 + l16;
    const float biasA = bias[nA];
    const float biasB = bias[nB];

    #pragma unroll
    for (int j = 0; j < 8; ++j) {
        const int mA = m0 + half * 8 + j;
        const int mB = mA + 16;
        out[(long long)mA * M_OUT + nA] = (float)c00[j] * s + biasA;
        out[(long long)mA * M_OUT + nB] = (float)c01[j] * s + biasB;
        out[(long long)mB * M_OUT + nA] = (float)c10[j] * s + biasA;
        out[(long long)mB * M_OUT + nB] = (float)c11[j] * s + biasB;
    }
}

extern "C" void kernel_launch(void* const* d_in, const int* in_sizes, int n_in,
                              void* d_out, int out_size, void* d_ws, size_t ws_size,
                              hipStream_t stream) {
    const float* x    = (const float*)d_in[0];
    const float* w    = (const float*)d_in[1];
    const float* bias = (const float*)d_in[2];
    float* out = (float*)d_out;

    const long long nx = in_sizes[0];
    const long long nw = in_sizes[1];
    const int nrows = (int)(nx / K_DIM);

    unsigned char* ws = (unsigned char*)d_ws;
    unsigned* slots = (unsigned*)ws;
    int* qw = (int*)(ws + 256);
    int* qx = (int*)(ws + 256 + nw);

    init_slots<<<1, 32, 0, stream>>>(slots);
    absmax_kernel<<<512, 256, 0, stream>>>(x, nx / 4, slots + 0);
    absmax_kernel<<<256, 256, 0, stream>>>(w, nw / 4, slots + 1);
    quantize_kernel<<<1024, 256, 0, stream>>>(x, qx, nx / 4, slots + 0);
    quantize_kernel<<<256,  256, 0, stream>>>(w, qw, nw / 4, slots + 1);

    dim3 grid(nrows / 128, M_OUT / 64);
    gemm_i8_wmma<<<grid, 256, 0, stream>>>(qx, qw, bias, out, slots);
}
